// MySeperableLayerNorm_81046032875553
// MI455X (gfx1250) — matrix operand order repaired and checked by reading, not compiled, not run on hardware
//
#include <hip/hip_runtime.h>
#include <hip/hip_bf16.h>
#include <math.h>

// ---------------- problem constants (from reference) ----------------
#define NROWS 200000
#define TD    480     // TOTAL_DIM
#define NS    128     // NUM_SCALAR
#define GIN_D 224     // GATE_IN
#define BND   112     // BN_DIM
#define NF    224     // NUM_FEATURES

// ---------------- tiling ----------------
#define WAVES 8
#define RPW   32            // rows per wave (two 16-row WMMA M-tiles)
#define NT1   7             // 112/16 N-tiles for GEMM1
#define KS1   7             // 224/32 K-steps for GEMM1
#define NT2   14            // 224/16 N-tiles for GEMM2
#define KS2   4             // 128/32 K-steps for GEMM2 (K=112 zero-padded)
#define GIN_STRIDE 232      // bf16 units; 464B rows: 16B aligned, conflict-free frag reads
#define H_STRIDE   136      // bf16 units; 272B rows: 16B aligned, conflict-free frag reads

typedef __attribute__((ext_vector_type(16))) __bf16 v16bf;
typedef __attribute__((ext_vector_type(8)))  __bf16 v8bf;
typedef __attribute__((ext_vector_type(8)))  float  v8f;
typedef __attribute__((ext_vector_type(4)))  float  v4f;

__device__ __forceinline__ unsigned short f2bf(float f) {
  unsigned int u = __builtin_bit_cast(unsigned int, f);
  u += 0x7FFFu + ((u >> 16) & 1u);          // round-to-nearest-even
  return (unsigned short)(u >> 16);
}

__device__ __forceinline__ v16bf combine16(v8bf lo, v8bf hi) {
  v16bf a;
#pragma unroll
  for (int i = 0; i < 8; ++i) { a[i] = lo[i]; a[i + 8] = hi[i]; }
  return a;
}

// fast sigmoid: v_exp_f32 + v_rcp_f32 (~1 ulp each) -- plenty for a gate
__device__ __forceinline__ float fast_sigmoid(float z) {
  return __builtin_amdgcn_rcpf(1.0f + __expf(-z));
}

// ---------------------------------------------------------------
// Prep: convert W1/W2 (f32, row-major [K][N]) into bf16 WMMA
// B-fragments in workspace.  B layout (wave32, 16x16x32 bf16):
//   lane l holds N = l&15, element e holds K = 16*(e>>3) + 8*(l>>4) + (e&7).
// Fragment = 32 contiguous bytes per lane, 1KB per fragment.
// ---------------------------------------------------------------
__global__ void prep_weights(const float* __restrict__ W1,
                             const float* __restrict__ W2,
                             unsigned short* __restrict__ ws) {
  const int n1 = NT1 * KS1 * 32 * 16;   // 25088 elems
  const int n2 = NT2 * KS2 * 32 * 16;   // 28672 elems
  int t = blockIdx.x * blockDim.x + threadIdx.x;
  if (t < n1) {
    int e = t & 15, lane = (t >> 4) & 31, f = t >> 9;
    int ks = f % KS1, nt = f / KS1;
    int k = 32 * ks + ((e >> 3) << 4) + ((lane >> 4) << 3) + (e & 7);
    int n = nt * 16 + (lane & 15);
    ws[t] = f2bf(W1[k * BND + n]);                       // k<224, n<112
  } else if (t < n1 + n2) {
    int u = t - n1;
    int e = u & 15, lane = (u >> 4) & 31, f = u >> 9;
    int ks = f & 3, nt = f >> 2;
    int k = 32 * ks + ((e >> 3) << 4) + ((lane >> 4) << 3) + (e & 7);
    int n = nt * 16 + (lane & 15);
    float v = (k < BND) ? W2[k * NF + n] : 0.0f;         // zero-pad K 112..127
    ws[n1 + u] = f2bf(v);
  }
}

// ---------------------------------------------------------------
// Main kernel: one wave = 32 rows (two 16-row WMMA tiles)
// ---------------------------------------------------------------
__global__ __launch_bounds__(WAVES * 32, 1)
void gate_kernel(const float* __restrict__ x,
                 const unsigned short* __restrict__ wfrag,
                 const float* __restrict__ b1,
                 const float* __restrict__ b2,
                 const float* __restrict__ affw,
                 const float* __restrict__ abias,
                 float* __restrict__ out) {
  __shared__ __align__(16) unsigned short gin[WAVES][RPW * GIN_STRIDE];
  __shared__ __align__(16) unsigned short hst[WAVES][16 * H_STRIDE];
  __shared__ float means[WAVES][RPW];

  const int wave = threadIdx.x >> 5;
  const int lane = threadIdx.x & 31;
  const int gw   = blockIdx.x * WAVES + wave;       // global wave id
  if (gw * RPW >= NROWS) return;                    // idle tail waves: no barriers used

  // ---------- pre-pass: lane owns one row ----------
  {
    const float* row = x + (size_t)(gw * RPW + lane) * TD;
    float s = 0.f;
#pragma unroll 8
    for (int j = 0; j < NS; j += 4) {
      v4f v = *(const v4f*)(row + j);
      s += v.x + v.y + v.z + v.w;
    }
    float mean = s * (1.0f / NS);
    means[wave][lane] = mean;
    unsigned short* g = &gin[wave][lane * GIN_STRIDE];
#pragma unroll 8
    for (int j = 0; j < NS; j += 4) {
      v4f v = *(const v4f*)(row + j);
      g[j + 0] = f2bf(v.x - mean);
      g[j + 1] = f2bf(v.y - mean);
      g[j + 2] = f2bf(v.z - mean);
      g[j + 3] = f2bf(v.w - mean);
    }
#pragma unroll 4
    for (int c = 0; c < 64; ++c) {                  // 64 channels of dim 3
      const float* p = row + NS + 3 * c;
      float a0 = p[0], a1 = p[1], a2 = p[2];
      g[NS + c] = f2bf(sqrtf((a0*a0 + a1*a1 + a2*a2) * (1.f/3.f) + 1e-6f));
    }
#pragma unroll 4
    for (int c = 0; c < 32; ++c) {                  // 32 channels of dim 5
      const float* p = row + NS + 192 + 5 * c;
      float ss = 0.f;
#pragma unroll
      for (int q = 0; q < 5; ++q) ss += p[q] * p[q];
      g[192 + c] = f2bf(sqrtf(ss * 0.2f + 1e-6f));
    }
  }
  asm volatile("s_wait_dscnt 0" ::: "memory");      // wave-local LDS visibility

  const unsigned short* w1f = wfrag;
  const unsigned short* w2f = wfrag + NT1 * KS1 * 512;
  const int r  = lane & 15;
  const int hi = lane >> 4;

#pragma unroll 1
  for (int t = 0; t < 2; ++t) {
    // ---- A-fragments of gate_in (bf16 A layout: {k..k+7, k+16..k+23} per lane) ----
    const unsigned short* ginT = &gin[wave][t * 16 * GIN_STRIDE];
    v16bf A1[KS1];
#pragma unroll
    for (int ks = 0; ks < KS1; ++ks) {
      const __bf16* base = (const __bf16*)(ginT + r * GIN_STRIDE + ks * 32 + hi * 8);
      A1[ks] = combine16(*(const v8bf*)base, *(const v8bf*)(base + 16));
    }

    // ---- GEMM1: h = silu(gate_in @ W1 + b1), staged bf16 in LDS ----
    unsigned short* h = hst[wave];
#pragma unroll 1
    for (int nt = 0; nt < NT1; ++nt) {              // rolled: cap register pressure
      v8f acc = {};
#pragma unroll
      for (int ks = 0; ks < KS1; ++ks) {
        const __bf16* bp = (const __bf16*)(w1f + ((nt * KS1 + ks) * 32 + lane) * 16);
        v16bf b = combine16(*(const v8bf*)bp, *(const v8bf*)(bp + 8));
        acc = __builtin_amdgcn_wmma_f32_16x16x32_bf16(
            false, A1[ks], false, b, (short)0, acc, false, false);
      }
      float bias = b1[nt * 16 + r];                 // D layout: lane -> N
#pragma unroll
      for (int v = 0; v < 8; ++v) {
        int m = v + 8 * hi;                         // D layout: VGPR -> M
        float z = acc[v] + bias;
        float sv = z * fast_sigmoid(z);             // silu
        h[m * H_STRIDE + nt * 16 + r] = f2bf(sv);
      }
    }
#pragma unroll
    for (int v = 0; v < 8; ++v)                     // zero K padding 112..127
      h[(v + 8 * hi) * H_STRIDE + BND + r] = 0;
    asm volatile("s_wait_dscnt 0" ::: "memory");

    // ---- A-fragments of h ----
    v16bf A2[KS2];
#pragma unroll
    for (int ks = 0; ks < KS2; ++ks) {
      const __bf16* base = (const __bf16*)(h + r * H_STRIDE + ks * 32 + hi * 8);
      A2[ks] = combine16(*(const v8bf*)base, *(const v8bf*)(base + 16));
    }

    // ---- GEMM2 + fused gating epilogue ----
    const int rowBase = gw * RPW + t * 16;
#pragma unroll 1
    for (int nt = 0; nt < NT2; ++nt) {              // rolled: cap register pressure
      v8f acc = {};
#pragma unroll
      for (int ks = 0; ks < KS2; ++ks) {
        const __bf16* bp = (const __bf16*)(w2f + ((nt * KS2 + ks) * 32 + lane) * 16);
        v16bf b = combine16(*(const v8bf*)bp, *(const v8bf*)(bp + 8));
        acc = __builtin_amdgcn_wmma_f32_16x16x32_bf16(
            false, A2[ks], false, b, (short)0, acc, false, false);
      }
      int n = nt * 16 + r;                          // gate channel
      float bb = b2[n], aw = affw[n];
#pragma unroll
      for (int v = 0; v < 8; ++v) {
        int m = v + 8 * hi;
        int rowi = rowBase + m;
        float z = acc[v] + bb;
        float gsc = 2.0f * aw * fast_sigmoid(z);    // 2*sigmoid*affine_weight
        const float* xr = x   + (size_t)rowi * TD;
        float*       yr = out + (size_t)rowi * TD;
        if (n < NS) {                               // scalar channel (uniform per nt)
          float mn = means[wave][t * 16 + m];
          yr[n] = (xr[n] - mn) * gsc + abias[n];
        } else if (n < 192) {                       // dim-3 vector channel
          int col = NS + 3 * (n - NS);
#pragma unroll
          for (int q = 0; q < 3; ++q) yr[col + q] = xr[col + q] * gsc;
        } else {                                    // dim-5 vector channel
          int col = 320 + 5 * (n - 192);
#pragma unroll
          for (int q = 0; q < 5; ++q) yr[col + q] = xr[col + q] * gsc;
        }
      }
    }
  }
}

extern "C" void kernel_launch(void* const* d_in, const int* in_sizes, int n_in,
                              void* d_out, int out_size, void* d_ws, size_t ws_size,
                              hipStream_t stream) {
  const float* x     = (const float*)d_in[0];
  const float* W1    = (const float*)d_in[1];
  const float* b1    = (const float*)d_in[2];
  const float* W2    = (const float*)d_in[3];
  const float* b2    = (const float*)d_in[4];
  const float* affw  = (const float*)d_in[5];
  const float* abias = (const float*)d_in[6];
  unsigned short* wfrag = (unsigned short*)d_ws;   // 107,520 B of fragments

  const int prepN = (NT1 * KS1 + NT2 * KS2) * 512; // 53,760 elements
  prep_weights<<<(prepN + 255) / 256, 256, 0, stream>>>(W1, W2, wfrag);

  const int totalWaves = NROWS / RPW;              // 6250 exactly
  const int nblocks = (totalWaves + WAVES - 1) / WAVES; // 782
  gate_kernel<<<nblocks, WAVES * 32, 0, stream>>>(x, wfrag, b1, b2, affw, abias,
                                                  (float*)d_out);
}
